// MultiHeadAttention_60155311948085
// MI455X (gfx1250) — compile-verified, matrix operand order checked by reading, not run on hardware
//
#include <hip/hip_runtime.h>

// ---------------------------------------------------------------------------
// MHA forward for MI455X (gfx1250, wave32).  All matmuls via
// v_wmma_f32_16x16x32_bf16.  Flash-attention inner loop: no S x S scores in
// HBM; K tiles staged with global_load_async_to_lds_b128 (ASYNCcnt).
// Workspace layout (needs 40 MB of d_ws):
//   [0,  8M)  Qp  bf16 [B*S, D]
//   [8M, 16M) Kp  bf16 [B*S, D]
//   [16M,24M) Vp  bf16 [B*S, D]
//   [24M,40M) AO  f32  [B*S, D]   (attention output, pre-Wo)
// ---------------------------------------------------------------------------

typedef __attribute__((ext_vector_type(16))) __bf16          v16bf;
typedef __attribute__((ext_vector_type(8)))  float           v8f;
typedef __attribute__((ext_vector_type(8)))  unsigned short  v8us;
typedef __attribute__((ext_vector_type(16))) unsigned short  v16us;

#define DMODEL 1024
#define SEQ    2048
#define NB     2
#define NH     16
#define DHEAD  64

__device__ __forceinline__ unsigned short f2bf(float f) {
  unsigned int u = __builtin_bit_cast(unsigned int, f);
  u += 0x7FFFu + ((u >> 16) & 1u);   // round-to-nearest-even
  return (unsigned short)(u >> 16);
}

__device__ __forceinline__ unsigned int pack2(float a, float b) {
  return (unsigned int)f2bf(a) | ((unsigned int)f2bf(b) << 16);
}

// Build one 16-element bf16 WMMA fragment for this lane.
// ISA 7.12.2 (16-bit A/B, 16x32): lane holds K = [kb..kb+7] and [kb+16..kb+23]
// relative to the 32-wide K-slice start; both halves are 16B-contiguous.
__device__ __forceinline__ v16bf frag16(const unsigned short* p) {
  v8us lo = *(const v8us*)(p);
  v8us hi = *(const v8us*)(p + 16);
  v16us t = __builtin_shufflevector(lo, hi,
      0, 1, 2, 3, 4, 5, 6, 7, 8, 9, 10, 11, 12, 13, 14, 15);
  return __builtin_bit_cast(v16bf, t);
}

// ---------------------------------------------------------------------------
// GEMM:  C[M,N] = A[M,K] (f32) * W[N,K]^T (f32, torch Linear layout) + bias[N]
// Block tile 128x256, BK=32 (= one bf16 WMMA K-depth), 8 waves as 2(M)x4(N),
// each wave 64x64 -> 4x4 f32 accumulators: 16 WMMA per 16 ds_load_b128.
// ---------------------------------------------------------------------------
#define BM  128
#define BN  256
#define BK  32
#define LDT 40   // padded LDS row (bf16 elems): 16B-aligned rows, no bank clash

template <bool OUT_BF16>
__global__ __launch_bounds__(256) void gemm_bt_bias(
    const float* __restrict__ A, const float* __restrict__ W,
    const float* __restrict__ bias, void* __restrict__ Cout,
    int M, int N, int K) {
  __shared__ unsigned short sA[BM * LDT];
  __shared__ unsigned short sB[BN * LDT];

  const int tid  = threadIdx.x;
  const int lane = tid & 31;
  const int wave = tid >> 5;
  const int wm   = wave >> 2;        // 0..1  (64-row band)
  const int wn   = wave & 3;         // 0..3  (64-col band)
  const int l16  = lane & 15;
  const int half = lane >> 4;
  const int kb   = half * 8;
  const int bm   = blockIdx.y * BM;
  const int bn   = blockIdx.x * BN;

  const int lrow = tid >> 1;          // 0..127: one A-tile row per thread pair
  const int lco4 = (tid & 1) * 4;     // float4 slot 0..3 or 4..7

  v8f zero = {};
  v8f acc[4][4];
#pragma unroll
  for (int mi = 0; mi < 4; ++mi)
#pragma unroll
    for (int ni = 0; ni < 4; ++ni) acc[mi][ni] = zero;

  for (int kt = 0; kt < K; kt += BK) {
    // A tile: 128 x 32 f32 -> bf16 LDS (two threads per row)
    const float* arow = A + (size_t)(bm + lrow) * K + kt;
#pragma unroll
    for (int j = 0; j < 4; ++j) {
      const int c4 = lco4 + j;
      float4 fa = *(const float4*)(arow + c4 * 4);
      uint2 pa;
      pa.x = pack2(fa.x, fa.y);
      pa.y = pack2(fa.z, fa.w);
      *(uint2*)&sA[lrow * LDT + c4 * 4] = pa;
    }
    // W tile: 256 x 32 f32 -> bf16 LDS (one thread per row)
    const float* wrow = W + (size_t)(bn + tid) * K + kt;
#pragma unroll
    for (int j = 0; j < 8; ++j) {
      float4 fw = *(const float4*)(wrow + j * 4);
      uint2 pw;
      pw.x = pack2(fw.x, fw.y);
      pw.y = pack2(fw.z, fw.w);
      *(uint2*)&sB[tid * LDT + j * 4] = pw;
    }
    if (kt + BK < K) {                       // global_prefetch_b8 next tiles
      __builtin_prefetch(arow + BK, 0, 1);
      __builtin_prefetch(wrow + BK, 0, 1);
    }
    __syncthreads();

    v16bf af[4];
#pragma unroll
    for (int mi = 0; mi < 4; ++mi)
      af[mi] = frag16(&sA[(wm * 64 + mi * 16 + l16) * LDT + kb]);
#pragma unroll
    for (int ni = 0; ni < 4; ++ni) {
      v16bf bfq = frag16(&sB[(wn * 64 + ni * 16 + l16) * LDT + kb]);
#pragma unroll
      for (int mi = 0; mi < 4; ++mi)
        acc[mi][ni] = __builtin_amdgcn_wmma_f32_16x16x32_bf16(
            false, af[mi], false, bfq, (short)0, acc[mi][ni], false, false);
    }
    __syncthreads();
  }

  // Epilogue per ISA C/D layout: VGPR r -> row (r + half*8), col = lane&15.
#pragma unroll
  for (int mi = 0; mi < 4; ++mi) {
#pragma unroll
    for (int ni = 0; ni < 4; ++ni) {
      const int col = bn + wn * 64 + ni * 16 + l16;
      const float bv = bias[col];
#pragma unroll
      for (int r = 0; r < 8; ++r) {
        const int row = bm + wm * 64 + mi * 16 + half * 8 + r;
        const float val = acc[mi][ni][r] + bv;
        if (OUT_BF16)
          ((unsigned short*)Cout)[(size_t)row * N + col] = f2bf(val);
        else
          ((float*)Cout)[(size_t)row * N + col] = val;
      }
    }
  }
}

// ---------------------------------------------------------------------------
// Flash attention: one block = 128 query rows of one (b, h).
// 8 waves, each owns 16 query rows; Q fragments stay in VGPRs.
// Per 64-key tile:
//   K staged row-major via GLOBAL_LOAD_ASYNC_TO_LDS_B128 (ASYNCcnt path),
//   V staged transposed ([d][k]) through VGPRs so PV B-fragments read 8
//   contiguous K-values, P re-laid-out through wave-private LDS.
// ---------------------------------------------------------------------------
#define AQ  128
#define AK  64
#define LKV 72   // padded LDS row (bf16 elems), 16B aligned

__global__ __launch_bounds__(256) void flash_attn(
    const unsigned short* __restrict__ Qp, const unsigned short* __restrict__ Kp,
    const unsigned short* __restrict__ Vp, const int* __restrict__ mask,
    float* __restrict__ Oout) {
  __shared__ unsigned short sK[AK * LKV];        // [key][d]
  __shared__ unsigned short sV[DHEAD * LKV];     // [d][key]  (transposed)
  __shared__ unsigned short sP[8 * 16 * LKV];    // per-wave 16 x 64 P tile

  const int tid  = threadIdx.x;
  const int lane = tid & 31;
  const int wave = tid >> 5;
  const int l16  = lane & 15;
  const int half = lane >> 4;
  const int kb   = half * 8;
  const int b    = blockIdx.y / NH;
  const int h    = blockIdx.y % NH;
  const int q0   = blockIdx.x * AQ;
  const size_t baseRow = (size_t)b * SEQ;
  const int hoff = h * DHEAD;

  // Resident Q fragments (A-matrix, M = query row, WMMA-K = d slice)
  v16bf qf[2];
  {
    const unsigned short* qptr =
        Qp + (baseRow + q0 + wave * 16 + l16) * DMODEL + hoff;
#pragma unroll
    for (int kk = 0; kk < 2; ++kk) qf[kk] = frag16(qptr + kk * 32 + kb);
  }

  v8f zero = {};
  v8f oacc[4];
#pragma unroll
  for (int ni = 0; ni < 4; ++ni) oacc[ni] = zero;
  float mrow[8], lrow[8];
#pragma unroll
  for (int r = 0; r < 8; ++r) { mrow[r] = -3.0e38f; lrow[r] = 0.0f; }

  const int qbase = q0 + wave * 16 + half * 8;   // global q row of VGPR r=0
  unsigned short* pbase = &sP[wave * 16 * LKV];

  for (int k0 = 0; k0 < SEQ; k0 += AK) {
    __syncthreads();   // protect sK/sV reuse across tiles
    // Stage K (async copy, row-major) and V (transposed) tiles: 64 x 64 bf16
#pragma unroll
    for (int it = 0; it < 2; ++it) {
      const int slot = it * 256 + tid;           // 0..511
      const int kr   = slot >> 3;                // key row 0..63
      const int c8   = slot & 7;                 // 8-elem column group
      const size_t g = (baseRow + k0 + kr) * DMODEL + hoff + c8 * 8;
      // K: global -> LDS direct (no VGPR round trip), tracked on ASYNCcnt
      {
        unsigned ldsK = (unsigned)(uintptr_t)&sK[kr * LKV + c8 * 8];
        unsigned long long gK = (unsigned long long)(uintptr_t)(Kp + g);
        asm volatile("global_load_async_to_lds_b128 %0, %1, off"
                     :: "v"(ldsK), "v"(gK) : "memory");
      }
      // V: needs element transpose -> through VGPRs
      v8us vv = *(const v8us*)(Vp + g);
#pragma unroll
      for (int j = 0; j < 8; ++j) sV[(c8 * 8 + j) * LKV + kr] = vv[j];
    }
    asm volatile("s_wait_asynccnt 0x0" ::: "memory");
    __syncthreads();

    // S = Q K^T  (WMMA-K = d, N = key index; K' rows are B-fragments as-is)
    v8f sacc[4];
#pragma unroll
    for (int ni = 0; ni < 4; ++ni) sacc[ni] = zero;
#pragma unroll
    for (int kk = 0; kk < 2; ++kk)
#pragma unroll
      for (int ni = 0; ni < 4; ++ni) {
        v16bf kf = frag16(&sK[(ni * 16 + l16) * LKV + kk * 32 + kb]);
        sacc[ni] = __builtin_amdgcn_wmma_f32_16x16x32_bf16(
            false, qf[kk], false, kf, (short)0, sacc[ni], false, false);
      }

    // scale 1/sqrt(64) + mask (coalesced b32 loads; mask lives in 192MB L2)
#pragma unroll
    for (int ni = 0; ni < 4; ++ni) {
      const int col = k0 + ni * 16 + l16;
      const int* mp = mask + ((size_t)b * SEQ + qbase) * SEQ + col;
#pragma unroll
      for (int r = 0; r < 8; ++r) {
        const float s = sacc[ni][r] * 0.125f;
        const int mv = mp[(size_t)r * SEQ];
        sacc[ni][r] = mv ? s : -1.0e9f;
      }
    }

    // Online softmax (rows of one half live on lanes 0-15 / 16-31)
    float alpha[8];
#pragma unroll
    for (int r = 0; r < 8; ++r) {
      float cm = fmaxf(fmaxf(sacc[0][r], sacc[1][r]),
                       fmaxf(sacc[2][r], sacc[3][r]));
#pragma unroll
      for (int off = 8; off >= 1; off >>= 1)
        cm = fmaxf(cm, __shfl_xor(cm, off, 32));
      const float nm = fmaxf(mrow[r], cm);
      alpha[r] = __expf(mrow[r] - nm);
      mrow[r] = nm;
    }
#pragma unroll
    for (int ni = 0; ni < 4; ++ni)
#pragma unroll
      for (int r = 0; r < 8; ++r)
        sacc[ni][r] = __expf(sacc[ni][r] - mrow[r]);
#pragma unroll
    for (int r = 0; r < 8; ++r) {
      float rs = (sacc[0][r] + sacc[1][r]) + (sacc[2][r] + sacc[3][r]);
#pragma unroll
      for (int off = 8; off >= 1; off >>= 1) rs += __shfl_xor(rs, off, 32);
      lrow[r] = lrow[r] * alpha[r] + rs;
    }
#pragma unroll
    for (int ni = 0; ni < 4; ++ni)
#pragma unroll
      for (int r = 0; r < 8; ++r) oacc[ni][r] *= alpha[r];

    // P (D-layout) -> bf16 -> wave-private LDS (A-fragment order)
#pragma unroll
    for (int ni = 0; ni < 4; ++ni)
#pragma unroll
      for (int r = 0; r < 8; ++r)
        pbase[(half * 8 + r) * LKV + ni * 16 + l16] = f2bf(sacc[ni][r]);

    // O += P V   (WMMA-K = key index, N = d; needs transposed sV)
#pragma unroll
    for (int kk = 0; kk < 2; ++kk) {
      v16bf pf = frag16(&pbase[l16 * LKV + kk * 32 + kb]);
#pragma unroll
      for (int ni = 0; ni < 4; ++ni) {
        v16bf vf = frag16(&sV[(ni * 16 + l16) * LKV + kk * 32 + kb]);
        oacc[ni] = __builtin_amdgcn_wmma_f32_16x16x32_bf16(
            false, pf, false, vf, (short)0, oacc[ni], false, false);
      }
    }
  }

  // Normalize and write attention output at [b*S+q, h*64+d] (== reference's
  // transpose(0,2,1,3).reshape(B,-1,D))
#pragma unroll
  for (int ni = 0; ni < 4; ++ni)
#pragma unroll
    for (int r = 0; r < 8; ++r) {
      const float val = oacc[ni][r] / lrow[r];
      Oout[(baseRow + qbase + r) * DMODEL + hoff + ni * 16 + l16] = val;
    }
}

// ---------------------------------------------------------------------------
extern "C" void kernel_launch(void* const* d_in, const int* in_sizes, int n_in,
                              void* d_out, int out_size, void* d_ws,
                              size_t ws_size, hipStream_t stream) {
  const float* q    = (const float*)d_in[0];
  const float* k    = (const float*)d_in[1];
  const float* v    = (const float*)d_in[2];
  const float* Wq   = (const float*)d_in[3];
  const float* bq   = (const float*)d_in[4];
  const float* Wk   = (const float*)d_in[5];
  const float* bk   = (const float*)d_in[6];
  const float* Wv   = (const float*)d_in[7];
  const float* bv   = (const float*)d_in[8];
  const float* Wo   = (const float*)d_in[9];
  const float* bo   = (const float*)d_in[10];
  const int*   mask = (const int*)d_in[11];

  char* ws = (char*)d_ws;
  unsigned short* Qp = (unsigned short*)(ws);
  unsigned short* Kp = (unsigned short*)(ws + (size_t)8  * 1024 * 1024);
  unsigned short* Vp = (unsigned short*)(ws + (size_t)16 * 1024 * 1024);
  float*          AO = (float*)         (ws + (size_t)24 * 1024 * 1024);

  const int M = NB * SEQ;   // 4096
  const dim3 ggrid(DMODEL / BN, M / BM);   // (4, 32)
  const dim3 gblk(256);

  gemm_bt_bias<true><<<ggrid, gblk, 0, stream>>>(q, Wq, bq, Qp, M, DMODEL, DMODEL);
  gemm_bt_bias<true><<<ggrid, gblk, 0, stream>>>(k, Wk, bk, Kp, M, DMODEL, DMODEL);
  gemm_bt_bias<true><<<ggrid, gblk, 0, stream>>>(v, Wv, bv, Vp, M, DMODEL, DMODEL);

  flash_attn<<<dim3(SEQ / AQ, NB * NH), gblk, 0, stream>>>(Qp, Kp, Vp, mask, AO);

  gemm_bt_bias<false><<<ggrid, gblk, 0, stream>>>(AO, Wo, bo, (float*)d_out, M,
                                                  DMODEL, DMODEL);
}
